// MambaStateSpaceLayer_55164559950115
// MI455X (gfx1250) — matrix operand order repaired and checked by reading, not compiled, not run on hardware
//
#include <hip/hip_runtime.h>
#include <hip/hip_bf16.h>

// ---------------------------------------------------------------------------
// Mamba SSM layer for MI455X (gfx1250, wave32, WMMA + async-to-LDS).
//   B=2, S=1024, D_MODEL=1024, D_INNER=2048, D_STATE=16, T=2048 tokens.
// GEMMs: bf16 WMMA (v_wmma_f32_16x16x32_bf16) with f32 accumulation, tiles
// staged with GLOBAL_LOAD_ASYNC_TO_LDS_B128 into 4-way-buffered LDS.
// ASYNCcnt retires in order per wave, so `s_wait_asynccnt 3` guarantees the
// previous stage landed while the newest stage (3 DMAs) stays in flight ->
// each tile DMA gets a full extra iteration of latency slack. The drain is
// peeled out of the loop so the steady state is branch-free (unrollable).
// Selective scan: chunked parallel scan (32 chunks x 32 steps).
// ---------------------------------------------------------------------------

#define B_SZ   2
#define SEQ    1024
#define DMODEL 1024
#define DINNER 2048
#define DSTATE 16
#define TOKENS (B_SZ * SEQ)   // 2048
#define NCHUNK 32
#define CLEN   (SEQ / NCHUNK) // 32

typedef __bf16 v16bf __attribute__((ext_vector_type(16)));
typedef float  v8f   __attribute__((ext_vector_type(8)));

// ------------------------- CDNA5 async-to-LDS helpers ----------------------
// DMA 16 bytes/lane from global memory straight into LDS (ASYNCcnt-tracked).
// LDS operand is the wave-relative LDS byte address = low 32 bits of the
// flat shared-aperture pointer (ISA 10.2).
__device__ __forceinline__ void async_copy_b128(const void* gsrc, void* lds_dst) {
  unsigned int loff = (unsigned int)(unsigned long long)lds_dst;
  unsigned long long gaddr = (unsigned long long)gsrc;
  asm volatile("global_load_async_to_lds_b128 %0, %1, off"
               :: "v"(loff), "v"(gaddr) : "memory");
}

__device__ __forceinline__ void wait_async_le3() {
#if __has_builtin(__builtin_amdgcn_s_wait_asynccnt)
  __builtin_amdgcn_s_wait_asynccnt(3);
#else
  asm volatile("s_wait_asynccnt 0x3" ::: "memory");
#endif
}
__device__ __forceinline__ void wait_async_le0() {
#if __has_builtin(__builtin_amdgcn_s_wait_asynccnt)
  __builtin_amdgcn_s_wait_asynccnt(0);
#else
  asm volatile("s_wait_asynccnt 0x0" ::: "memory");
#endif
}

// ------------------------------ f32 -> bf16 --------------------------------
__global__ void cvt_f32_bf16_v4(const float* __restrict__ src,
                                __bf16* __restrict__ dst, int n4) {
  int i = blockIdx.x * blockDim.x + threadIdx.x;
  if (i < n4) {
    float4 f = ((const float4*)src)[i];
    union { __bf16 h[4]; uint2 u; } r;
    r.h[0] = (__bf16)f.x; r.h[1] = (__bf16)f.y;
    r.h[2] = (__bf16)f.z; r.h[3] = (__bf16)f.w;
    *(uint2*)&dst[(size_t)i * 4] = r.u;
  }
}

// ------------------------------- WMMA GEMM ---------------------------------
// O[M,N] = act( A[M,K](bf16) * W[N,K](bf16)^T + bias[N] ), f32 output.
// Block tile 64(M) x 128(N), K-step 32; 8 waves: wave%4 -> 16-row strip,
// wave/4 -> 64-col strip (4 WMMA subtiles, A-fragment reused 4x).
// 4-way buffered LDS (power-of-two index), DMA issued two tiles ahead.
#define TM 64
#define TN 128
#define TK 32
#define SK (TK + 8)   // +16B pad per row: keeps b128 alignment, spreads banks
#define NBUF 4

union Frag16 { uint4 u[2]; v16bf v; };

__global__ __launch_bounds__(256)
void gemm_bf16_wmma(const __bf16* __restrict__ A,   // [M,K] row-major
                    const __bf16* __restrict__ W,   // [N,K] row-major
                    const float*  __restrict__ bias,// [N]
                    float*        __restrict__ O,   // [M,N]
                    int M, int N, int K,
                    int act_mode, int act_from_col) // 0=none 1=relu(col>=c) 2=sigmoid
{
  __shared__ __bf16 As[NBUF][TM][SK];
  __shared__ __bf16 Bs[NBUF][TN][SK];

  const int bn = blockIdx.x * TN;
  const int bm = blockIdx.y * TM;
  const int tid  = threadIdx.x;
  const int wave = tid >> 5;
  const int lane = tid & 31;
  const int wm   = (wave & 3) << 4;   // 0,16,32,48
  const int wn   = (wave >> 2) << 6;  // 0,64
  const int half = lane >> 4;         // K half select (ISA 16-bit A layout)
  const int l16  = lane & 15;

  // per-thread staging coordinates (3 x b128 per thread per K-step)
  const int rA = tid >> 2, cA = (tid & 3) * 8;
  const int rB = tid >> 1, cB = (tid & 1) * 16;

  v8f acc[4] = {v8f{}, v8f{}, v8f{}, v8f{}};

  auto stage = [&](int buf, int k0) {
    async_copy_b128(&A[(size_t)(bm + rA) * K + k0 + cA], &As[buf][rA][cA]);
    const __bf16* s = &W[(size_t)(bn + rB) * K + k0 + cB];
    async_copy_b128(&s[0], &Bs[buf][rB][cB]);
    async_copy_b128(&s[8], &Bs[buf][rB][cB + 8]);
  };

  // A fragment (16x32 bf16): lane<16 -> M=lane, K 0..7 & 16..23;
  // lane>=16 -> M=lane-16, K 8..15 & 24..31  (ISA 7.12.2).
  // B fragment (32x16 bf16): N = lane&15, same K order as A fragment.
  auto compute = [&](int buf) {
    Frag16 af;
    af.u[0] = *(const uint4*)&As[buf][wm + l16][half * 8];
    af.u[1] = *(const uint4*)&As[buf][wm + l16][16 + half * 8];
#pragma unroll
    for (int t = 0; t < 4; ++t) {
      Frag16 bf;
      bf.u[0] = *(const uint4*)&Bs[buf][wn + t * 16 + l16][half * 8];
      bf.u[1] = *(const uint4*)&Bs[buf][wn + t * 16 + l16][16 + half * 8];
      acc[t] = __builtin_amdgcn_wmma_f32_16x16x32_bf16(
          false, af.v, false, bf.v, (short)0, acc[t], false, false);
    }
  };

  const int nk = K / TK;                // >= 32 for all our shapes
  stage(0, 0);
  stage(1, TK);
  wait_async_le3();                     // stage 0 landed, stage 1 in flight
  __syncthreads();

  // Steady state: branch-free body -> unrollable; &3 buffer indexing.
  int kt = 0;
  for (; kt < nk - 2; ++kt) {
    stage((kt + 2) & (NBUF - 1), (kt + 2) * TK);  // DMA 2 tiles ahead
    compute(kt & (NBUF - 1));
    wait_async_le3();   // in-order: retires the DMA for tile kt+1
    __syncthreads();    // everyone done reading tile kt
  }
  // Drain: tile nk-2 (already resident), then tile nk-1.
  compute(kt & (NBUF - 1));
  wait_async_le0();     // last stage (tile nk-1) fully landed
  __syncthreads();
  ++kt;
  compute(kt & (NBUF - 1));   // no barrier needed: epilogue reads own acc only

  // Epilogue: C/D layout — VGPR r holds M = r + 8*(lane>=16), N = lane&15.
  const int rbase = half * 8;
#pragma unroll
  for (int t = 0; t < 4; ++t) {
    int col = bn + wn + t * 16 + l16;
    float bv = bias[col];
#pragma unroll
    for (int r = 0; r < 8; ++r) {
      int row = bm + wm + rbase + r;
      float v = acc[t][r] + bv;
      if (act_mode == 1) { if (col >= act_from_col && v < 0.0f) v = 0.0f; }
      else if (act_mode == 2) { v = 1.0f / (1.0f + __expf(-v)); }
      O[(size_t)row * N + col] = v;
    }
  }
}

// ------------------------- selective scan (chunked) ------------------------
// h_t = h_{t-1} * exp(A*dt_t) + (x_t*dt_t)*B   (per d-channel, 16 states)
// Pass 1: per (b,d,chunk) compute P = prod(exp(A*dt)) and V = h(chunk, h0=0).
__global__ __launch_bounds__(256)
void scan_pass1(const float* __restrict__ proj,   // [T, 2*DINNER] x_proj|delta
                const float* __restrict__ A_log,  // [DINNER, DSTATE]
                const float* __restrict__ B_mat,  // [DINNER, DSTATE]
                float* __restrict__ P, float* __restrict__ V)
{
  int idx = blockIdx.x * blockDim.x + threadIdx.x;    // 131072 threads
  int d = idx & (DINNER - 1);
  int b = (idx >> 11) & (B_SZ - 1);
  int c = idx >> 12;

  float a[DSTATE], Bv[DSTATE], h[DSTATE], p[DSTATE];
#pragma unroll
  for (int n = 0; n < DSTATE; ++n) {
    a[n]  = -__expf(A_log[d * DSTATE + n]);
    Bv[n] = B_mat[d * DSTATE + n];
    h[n] = 0.0f; p[n] = 1.0f;
  }

  for (int s = 0; s < CLEN; ++s) {
    int tt = b * SEQ + c * CLEN + s;
    size_t row = (size_t)tt * (2 * DINNER);
    float xv = proj[row + d];
    float dt = proj[row + DINNER + d];   // already ReLU'd in GEMM epilogue
    float u  = xv * dt;
#pragma unroll
    for (int n = 0; n < DSTATE; ++n) {
      float e = __expf(a[n] * dt);
      p[n] *= e;
      h[n] = h[n] * e + u * Bv[n];
    }
  }
  size_t base = (((size_t)c * B_SZ + b) * DINNER + d) * DSTATE;
#pragma unroll
  for (int n = 0; n < DSTATE; ++n) { P[base + n] = p[n]; V[base + n] = h[n]; }
}

// Pass 2: serial composition over the 32 chunks per (b,d,n); emits the
// incoming state Hin for every chunk.
__global__ __launch_bounds__(256)
void scan_pass2(const float* __restrict__ P, const float* __restrict__ V,
                float* __restrict__ Hin)
{
  int idx = blockIdx.x * blockDim.x + threadIdx.x;    // 65536 threads
  int n = idx & (DSTATE - 1);
  int d = (idx >> 4) & (DINNER - 1);
  int b = (idx >> 15) & (B_SZ - 1);
  float h = 0.0f;
  for (int c = 0; c < NCHUNK; ++c) {
    size_t off = (((size_t)c * B_SZ + b) * DINNER + d) * DSTATE + n;
    Hin[off] = h;
    h = P[off] * h + V[off];
  }
}

// Pass 3: replay each chunk from Hin, emit gated output in bf16 for the
// final WMMA GEMM:  gy[t,d] = gate * (h.c0 + x_proj*Dv)
__global__ __launch_bounds__(256)
void scan_pass3(const float* __restrict__ proj,
                const float* __restrict__ A_log,
                const float* __restrict__ B_mat,
                const float* __restrict__ C_mat,   // row 0 used
                const float* __restrict__ D_vec,
                const float* __restrict__ gate,    // [T, DINNER]
                const float* __restrict__ Hin,
                __bf16* __restrict__ gy)           // [T, DINNER]
{
  int idx = blockIdx.x * blockDim.x + threadIdx.x;    // 131072 threads
  int d = idx & (DINNER - 1);
  int b = (idx >> 11) & (B_SZ - 1);
  int c = idx >> 12;

  float a[DSTATE], Bv[DSTATE], c0[DSTATE], h[DSTATE];
  size_t base = (((size_t)c * B_SZ + b) * DINNER + d) * DSTATE;
#pragma unroll
  for (int n = 0; n < DSTATE; ++n) {
    a[n]  = -__expf(A_log[d * DSTATE + n]);
    Bv[n] = B_mat[d * DSTATE + n];
    c0[n] = C_mat[n];                     // C_mat[0][n]
    h[n]  = Hin[base + n];
  }
  float Dv = D_vec[d];

  for (int s = 0; s < CLEN; ++s) {
    int tt = b * SEQ + c * CLEN + s;
    size_t row = (size_t)tt * (2 * DINNER);
    float xv = proj[row + d];
    float dt = proj[row + DINNER + d];
    float u  = xv * dt;
    float y = 0.0f;
#pragma unroll
    for (int n = 0; n < DSTATE; ++n) {
      float e = __expf(a[n] * dt);
      h[n] = h[n] * e + u * Bv[n];
      y += h[n] * c0[n];
    }
    float g = gate[(size_t)tt * DINNER + d];
    gy[(size_t)tt * DINNER + d] = (__bf16)(g * (y + xv * Dv));
  }
}

// ------------------------------- launcher ----------------------------------
extern "C" void kernel_launch(void* const* d_in, const int* in_sizes, int n_in,
                              void* d_out, int out_size, void* d_ws, size_t ws_size,
                              hipStream_t stream) {
  const float* x      = (const float*)d_in[0];   // [2,1024,1024]
  const float* W_in   = (const float*)d_in[1];   // [4096,1024]
  const float* b_in   = (const float*)d_in[2];
  const float* W_gate = (const float*)d_in[3];   // [2048,1024]
  const float* b_gate = (const float*)d_in[4];
  const float* W_out  = (const float*)d_in[5];   // [1024,2048]
  const float* b_out  = (const float*)d_in[6];
  const float* A_log  = (const float*)d_in[7];   // [2048,16]
  const float* B_mat  = (const float*)d_in[8];
  const float* C_mat  = (const float*)d_in[9];
  const float* D_vec  = (const float*)d_in[10];
  float* out = (float*)d_out;

  // workspace carve-out (256B aligned slabs)
  size_t off = 0;
  auto carve = [&](size_t bytes) -> void* {
    void* p = (char*)d_ws + off;
    off += (bytes + 255) & ~(size_t)255;
    return p;
  };
  __bf16* xb   = (__bf16*)carve((size_t)TOKENS * DMODEL * 2);
  __bf16* Winb = (__bf16*)carve((size_t)2 * DINNER * DMODEL * 2);
  __bf16* Wgb  = (__bf16*)carve((size_t)DINNER * DMODEL * 2);
  __bf16* Wob  = (__bf16*)carve((size_t)DMODEL * DINNER * 2);
  float*  proj = (float*) carve((size_t)TOKENS * 2 * DINNER * 4);
  float*  gate = (float*) carve((size_t)TOKENS * DINNER * 4);
  float*  Pbuf = (float*) carve((size_t)NCHUNK * B_SZ * DINNER * DSTATE * 4);
  float*  Vbuf = (float*) carve((size_t)NCHUNK * B_SZ * DINNER * DSTATE * 4);
  float*  Hin  = (float*) carve((size_t)NCHUNK * B_SZ * DINNER * DSTATE * 4);
  __bf16* gy   = (__bf16*)carve((size_t)TOKENS * DINNER * 2);
  (void)ws_size;

  // 1) down-convert activations + weights to bf16 (4 elems/thread)
  auto cvt = [&](const float* s, __bf16* dp, int n) {
    int n4 = n / 4;
    cvt_f32_bf16_v4<<<(n4 + 255) / 256, 256, 0, stream>>>(s, dp, n4);
  };
  cvt(x,      xb,   TOKENS * DMODEL);
  cvt(W_in,   Winb, 2 * DINNER * DMODEL);
  cvt(W_gate, Wgb,  DINNER * DMODEL);
  cvt(W_out,  Wob,  DMODEL * DINNER);

  // 2) proj = x @ W_in.T + b_in ; cols >= DINNER get ReLU (delta)
  {
    dim3 g((2 * DINNER) / TN, TOKENS / TM);
    gemm_bf16_wmma<<<g, 256, 0, stream>>>(xb, Winb, b_in, proj,
                                          TOKENS, 2 * DINNER, DMODEL,
                                          /*act*/1, /*from*/DINNER);
  }
  // 3) gate = sigmoid(x @ W_gate.T + b_gate)
  {
    dim3 g(DINNER / TN, TOKENS / TM);
    gemm_bf16_wmma<<<g, 256, 0, stream>>>(xb, Wgb, b_gate, gate,
                                          TOKENS, DINNER, DMODEL,
                                          /*act*/2, 0);
  }
  // 4) chunked parallel selective scan
  {
    int t1 = B_SZ * DINNER * NCHUNK;          // 131072
    scan_pass1<<<t1 / 256, 256, 0, stream>>>(proj, A_log, B_mat, Pbuf, Vbuf);
    int t2 = B_SZ * DINNER * DSTATE;          // 65536
    scan_pass2<<<t2 / 256, 256, 0, stream>>>(Pbuf, Vbuf, Hin);
    scan_pass3<<<t1 / 256, 256, 0, stream>>>(proj, A_log, B_mat, C_mat, D_vec,
                                             gate, Hin, gy);
  }
  // 5) out = gy @ W_out.T + b_out
  {
    dim3 g(DMODEL / TN, TOKENS / TM);
    gemm_bf16_wmma<<<g, 256, 0, stream>>>(gy, Wob, b_out, out,
                                          TOKENS, DMODEL, DINNER,
                                          /*act*/0, 0);
  }
}
